// MambaEncoder_29592324670026
// MI455X (gfx1250) — compile-verified, matrix operand order checked by reading, not run on hardware
//
#include <hip/hip_runtime.h>
#include <hip/hip_bf16.h>

typedef __attribute__((ext_vector_type(16))) __bf16 v16bf;
typedef __attribute__((ext_vector_type(8)))  float  v8f;

static __device__ __forceinline__ unsigned short bf_from_f32(float f) {
  __hip_bfloat16 h = __float2bfloat16(f);
  return __builtin_bit_cast(unsigned short, h);
}
static __device__ __forceinline__ float softplus_f(float v) {
  return (v > 20.f) ? v : log1pf(__expf(v));
}
static __device__ __forceinline__ float gelu_f(float v) {
  return 0.5f * v * (1.f + erff(v * 0.70710678118654752f));
}
static __device__ __forceinline__ float silu_f(float v) {
  return v / (1.f + __expf(-v));
}

// ---------------------------------------------------------------------------
// bf16 GEMM:  C[M,N] = act( A[M,K](lda) * B[N,K]^T + bias[N] )
// Host guarantees: K multiple of 32 (dt_w zero-padded), lda multiple of 8;
// A/B live in the workspace arena so tile overreads past M/N are harmless
// (lanes with n >= N never store; k-range is exact).
// Block: 256 thr = 8 waves; tile 256(M) x 32(N); each wave computes a 32x32
// patch = 4 x v_wmma_f32_16x16x32_bf16 per K-step from 2 A + 2 B fragments.
// Staging uses CDNA5 async global->LDS DMA (no VGPR round-trip, ASYNCcnt).
// ---------------------------------------------------------------------------
#define GTM 256
#define GTN 32
#define GTK 32
#define GLD (GTK + 8)   // 40 bf16 = 80B row stride: 16B aligned, conflict-free

__global__ __launch_bounds__(256) void gemm_bf16_kernel(
    const unsigned short* __restrict__ A, int lda,
    const unsigned short* __restrict__ B,      // N x K row-major (ldb = K)
    const float* __restrict__ bias,            // nullable
    float* __restrict__ Cf,                    // nullable
    unsigned short* __restrict__ Cb,           // nullable
    int M, int N, int K, int ldc, int act, int transC) {
  __shared__ __align__(16) __bf16 As[GTM][GLD];
  __shared__ __align__(16) __bf16 Bs[GTN][GLD];
  const int tid  = threadIdx.x;
  const int wave = tid >> 5;
  const int lane = tid & 31;
  const int m0 = blockIdx.x * GTM;
  const int n0 = blockIdx.y * GTN;

  // A staging: thread -> rows rbase+{0,64,128,192}, one 16B chunk per row
  const int chunk = (tid & 3) * 8;       // bf16 col offset within row
  const int rbase = tid >> 2;            // 0..63
  // B staging: all 256 threads, 32 rows x 8 x 8B chunks
  const int brow = tid >> 3;             // 0..31
  const int bcol = (tid & 7) * 4;        // bf16 col offset (8B chunks)

  const unsigned long long gA = (unsigned long long)(A + (long)(m0 + rbase) * lda + chunk);
  const unsigned long long stepA = (unsigned long long)(64L * lda * 2);
  const unsigned long long gB = (unsigned long long)(B + (long)(n0 + brow) * K + bcol);
  const unsigned ldsA0 = (unsigned)(unsigned long long)&As[rbase +   0][chunk];
  const unsigned ldsA1 = (unsigned)(unsigned long long)&As[rbase +  64][chunk];
  const unsigned ldsA2 = (unsigned)(unsigned long long)&As[rbase + 128][chunk];
  const unsigned ldsA3 = (unsigned)(unsigned long long)&As[rbase + 192][chunk];
  const unsigned ldsB  = (unsigned)(unsigned long long)&Bs[brow][bcol];

  v8f acc00 = {}, acc01 = {}, acc10 = {}, acc11 = {};
  for (int k0 = 0; k0 < K; k0 += GTK) {
    const unsigned long long koff = (unsigned long long)(2 * k0);
    // CDNA5 async global->LDS staging: no VGPR round-trip, tracked by ASYNCcnt
    asm volatile("global_load_async_to_lds_b128 %0, %1, off"
                 :: "v"(ldsA0), "v"(gA + koff) : "memory");
    asm volatile("global_load_async_to_lds_b128 %0, %1, off"
                 :: "v"(ldsA1), "v"(gA + stepA + koff) : "memory");
    asm volatile("global_load_async_to_lds_b128 %0, %1, off"
                 :: "v"(ldsA2), "v"(gA + 2 * stepA + koff) : "memory");
    asm volatile("global_load_async_to_lds_b128 %0, %1, off"
                 :: "v"(ldsA3), "v"(gA + 3 * stepA + koff) : "memory");
    asm volatile("global_load_async_to_lds_b64 %0, %1, off"
                 :: "v"(ldsB), "v"(gB + koff) : "memory");
    asm volatile("s_wait_asynccnt 0x0" ::: "memory");
    __syncthreads();

    // Fragment loads per CDNA5 16-bit layout: lane r=row, kb selects K half
    const int r  = lane & 15;
    const int kb = (lane >> 4) * 8;
    v16bf af0, af1, bf0, bf1;
#pragma unroll
    for (int e = 0; e < 8; ++e) {
      af0[e]     = As[wave * 32 + r][kb + e];
      af0[e + 8] = As[wave * 32 + r][16 + kb + e];
      af1[e]     = As[wave * 32 + 16 + r][kb + e];
      af1[e + 8] = As[wave * 32 + 16 + r][16 + kb + e];
      bf0[e]     = Bs[r][kb + e];
      bf0[e + 8] = Bs[r][16 + kb + e];
      bf1[e]     = Bs[16 + r][kb + e];
      bf1[e + 8] = Bs[16 + r][16 + kb + e];
    }
    acc00 = __builtin_amdgcn_wmma_f32_16x16x32_bf16(false, af0, false, bf0,
                                                    (short)0, acc00, false, false);
    acc01 = __builtin_amdgcn_wmma_f32_16x16x32_bf16(false, af0, false, bf1,
                                                    (short)0, acc01, false, false);
    acc10 = __builtin_amdgcn_wmma_f32_16x16x32_bf16(false, af1, false, bf0,
                                                    (short)0, acc10, false, false);
    acc11 = __builtin_amdgcn_wmma_f32_16x16x32_bf16(false, af1, false, bf1,
                                                    (short)0, acc11, false, false);
    __syncthreads();
  }

  // Epilogue. C layout: VGPR r: lanes 0-15 -> M=r, lanes 16-31 -> M=r+8
#pragma unroll
  for (int ni = 0; ni < 2; ++ni) {
    const int n = n0 + ni * 16 + (lane & 15);
    if (n >= N) continue;
    const float bv = bias ? bias[n] : 0.f;
#pragma unroll
    for (int mi = 0; mi < 2; ++mi) {
      const v8f* ap = (mi == 0) ? (ni == 0 ? &acc00 : &acc01)
                                : (ni == 0 ? &acc10 : &acc11);
#pragma unroll
      for (int r = 0; r < 8; ++r) {
        int m = m0 + wave * 32 + mi * 16 + ((lane < 16) ? r : r + 8);
        if (m < M) {
          float v = (*ap)[r] + bv;
          if (act == 1)      v = softplus_f(v);
          else if (act == 2) v = gelu_f(v);
          if (Cf) {
            if (transC) Cf[(long)n * M + m] = v;
            else        Cf[(long)m * ldc + n] = v;
          }
          if (Cb) Cb[(long)m * ldc + n] = bf_from_f32(v);
        }
      }
    }
  }
}

// ---------------------------------------------------------------------------
// Stem conv: x (1,1,64^3) -> h (64, 32^3); 7^3, stride 2, pad 3
// ---------------------------------------------------------------------------
__global__ void stem_kernel(const float* __restrict__ x,
                            const float* __restrict__ w,   // (64, 343)
                            const float* __restrict__ b,
                            float* __restrict__ out) {
  const long L = 32L * 32 * 32;
  long idx = (long)blockIdx.x * blockDim.x + threadIdx.x;
  if (idx >= 64 * L) return;
  int c = (int)(idx / L);
  long pl = idx % L;
  int oz = (int)(pl >> 10), oy = (int)((pl >> 5) & 31), ox = (int)(pl & 31);
  float acc = b[c];
  const float* wp = w + (long)c * 343;
  int k = 0;
  for (int kz = 0; kz < 7; ++kz) {
    int iz = 2 * oz - 3 + kz;
    for (int ky = 0; ky < 7; ++ky) {
      int iy = 2 * oy - 3 + ky;
      for (int kx = 0; kx < 7; ++kx, ++k) {
        int ix = 2 * ox - 3 + kx;
        if ((unsigned)iz < 64u && (unsigned)iy < 64u && (unsigned)ix < 64u)
          acc += wp[k] * x[((long)iz * 64 + iy) * 64 + ix];
      }
    }
  }
  out[idx] = acc;
}

// ---------------------------------------------------------------------------
// Per-channel mean / rstd over spatial (instance norm stats, eps = 1e-5)
// ---------------------------------------------------------------------------
__global__ __launch_bounds__(256) void chan_stats_kernel(
    const float* __restrict__ h, float* __restrict__ mean,
    float* __restrict__ rstd, int L) {
  int c = blockIdx.x;
  const float* p = h + (long)c * L;
  float s = 0.f, s2 = 0.f;
  for (int i = threadIdx.x; i < L; i += blockDim.x) {
    float v = p[i]; s += v; s2 += v * v;
  }
  __shared__ float sh[256], sh2[256];
  sh[threadIdx.x] = s; sh2[threadIdx.x] = s2;
  __syncthreads();
  for (int o = 128; o > 0; o >>= 1) {
    if (threadIdx.x < o) { sh[threadIdx.x] += sh[threadIdx.x + o];
                           sh2[threadIdx.x] += sh2[threadIdx.x + o]; }
    __syncthreads();
  }
  if (threadIdx.x == 0) {
    float m = sh[0] / L;
    float v = sh2[0] / L - m * m;
    if (v < 0.f) v = 0.f;
    mean[c] = m;
    rstd[c] = rsqrtf(v + 1e-5f);
  }
}

// Instance-normalized transpose (C,L) -> bf16 (L,C) for MLP GEMM input
__global__ void in_transpose_kernel(const float* __restrict__ h,
                                    const float* __restrict__ mean,
                                    const float* __restrict__ rstd,
                                    unsigned short* __restrict__ out,
                                    int C, int L) {
  long idx = (long)blockIdx.x * blockDim.x + threadIdx.x;
  long total = (long)C * L;
  if (idx >= total) return;
  int c = (int)(idx / L);
  long l = idx % L;
  float v = (h[idx] - mean[c]) * rstd[c];
  out[l * C + c] = bf_from_f32(v);
}

// ---------------------------------------------------------------------------
// Down conv: instance_norm(h (C1,S1^3)) -> conv 2^3 stride 2 -> (C2,(S1/2)^3)
// ---------------------------------------------------------------------------
__global__ void down_conv_kernel(const float* __restrict__ h,
                                 const float* __restrict__ mean,
                                 const float* __restrict__ rstd,
                                 const float* __restrict__ w,  // (C2,C1,2,2,2)
                                 const float* __restrict__ b,
                                 float* __restrict__ out,
                                 int C1, int S1, int C2) {
  int S2 = S1 >> 1;
  long L2 = (long)S2 * S2 * S2;
  long idx = (long)blockIdx.x * blockDim.x + threadIdx.x;
  if (idx >= (long)C2 * L2) return;
  int c2 = (int)(idx / L2);
  long pl = idx % L2;
  int oz = (int)(pl / (S2 * S2));
  int rem = (int)(pl % (S2 * S2));
  int oy = rem / S2, ox = rem % S2;
  float acc = b[c2];
  for (int c1 = 0; c1 < C1; ++c1) {
    float m = mean[c1], r = rstd[c1];
    const float* wp = w + ((long)c2 * C1 + c1) * 8;
    const float* hp = h + (long)c1 * S1 * S1 * S1;
#pragma unroll
    for (int dz = 0; dz < 2; ++dz)
#pragma unroll
      for (int dy = 0; dy < 2; ++dy)
#pragma unroll
        for (int dx = 0; dx < 2; ++dx) {
          long ii = ((long)(2 * oz + dz) * S1 + (2 * oy + dy)) * S1 + (2 * ox + dx);
          acc += wp[dz * 4 + dy * 2 + dx] * (hp[ii] - m) * r;
        }
  }
  out[idx] = acc;
}

// ---------------------------------------------------------------------------
// Layer norm over channels: h (C,L) f32 -> bf16 (L,C), eps = 1e-6
// ---------------------------------------------------------------------------
__global__ void layernorm_kernel(const float* __restrict__ h,
                                 const float* __restrict__ lw,
                                 const float* __restrict__ lb,
                                 unsigned short* __restrict__ out,
                                 int C, int L) {
  int l = blockIdx.x * blockDim.x + threadIdx.x;
  if (l >= L) return;
  float s = 0.f, s2 = 0.f;
  for (int c = 0; c < C; ++c) {
    float v = h[(long)c * L + l];
    s += v; s2 += v * v;
  }
  float m = s / C;
  float var = s2 / C - m * m;
  if (var < 0.f) var = 0.f;
  float r = rsqrtf(var + 1e-6f);
  for (int c = 0; c < C; ++c) {
    float v = (h[(long)c * L + l] - m) * r * lw[c] + lb[c];
    out[(long)l * C + c] = bf_from_f32(v);
  }
}

// ---------------------------------------------------------------------------
// Depthwise causal conv1d (k=4) + bias + SiLU on xin = xz[:, 0:di]
// ---------------------------------------------------------------------------
__global__ void dwconv_silu_kernel(const float* __restrict__ xz,
                                   const float* __restrict__ w,  // (di,4)
                                   const float* __restrict__ b,
                                   float* __restrict__ xc,
                                   unsigned short* __restrict__ xcb,
                                   int L, int di) {
  long idx = (long)blockIdx.x * blockDim.x + threadIdx.x;
  long total = (long)L * di;
  if (idx >= total) return;
  int e = (int)(idx % di);
  long t = idx / di;
  float acc = b[e];
#pragma unroll
  for (int j = 0; j < 4; ++j) {
    long tt = t - 3 + j;
    if (tt >= 0) acc += w[e * 4 + j] * xz[tt * (2L * di) + e];
  }
  float s = silu_f(acc);
  xc[idx] = s;
  xcb[idx] = bf_from_f32(s);
}

// ---------------------------------------------------------------------------
// Chunked selective scan (2-pass linear-recurrence parallelization)
// ---------------------------------------------------------------------------
__global__ void scan_pass_kernel(const float* __restrict__ dt,
                                 const float* __restrict__ xdbl,  // (L,fp)
                                 const float* __restrict__ xc,
                                 const float* __restrict__ xz,    // (L,2di)
                                 const float* __restrict__ A_log, // (di,16)
                                 const float* __restrict__ Dp,
                                 float* __restrict__ P,
                                 float* __restrict__ Hend,
                                 const float* __restrict__ Hinit,
                                 unsigned short* __restrict__ ybf,
                                 int L, int di, int fp, int dtr,
                                 int NC, int CL, int pass) {
  int idx = blockIdx.x * blockDim.x + threadIdx.x;
  if (idx >= NC * di) return;
  int e = idx % di;
  int ci = idx / di;
  float A[16], h[16], p[16];
#pragma unroll
  for (int s = 0; s < 16; ++s) {
    A[s] = -__expf(A_log[e * 16 + s]);
    p[s] = 1.f;
    h[s] = (pass == 2 && Hinit) ? Hinit[((long)(e * 16 + s)) * NC + ci] : 0.f;
  }
  float Dv = Dp[e];
  long t0 = (long)ci * CL;
  long t1 = t0 + CL; if (t1 > L) t1 = L;
  for (long t = t0; t < t1; ++t) {
    float dtv = dt[t * di + e];
    float xcv = xc[t * di + e];
    const float* Bm = xdbl + t * fp + dtr;
    const float* Cm = Bm + 16;
    float y = 0.f;
#pragma unroll
    for (int s = 0; s < 16; ++s) {
      float a = __expf(A[s] * dtv);
      h[s] = a * h[s] + dtv * Bm[s] * xcv;
      if (pass == 1) p[s] *= a;
      else           y += h[s] * Cm[s];
    }
    if (pass == 2) {
      y += xcv * Dv;
      float z = xz[t * (2L * di) + di + e];
      y *= silu_f(z);
      ybf[t * di + e] = bf_from_f32(y);
    }
  }
  if (pass == 1) {
#pragma unroll
    for (int s = 0; s < 16; ++s) {
      long o = ((long)(e * 16 + s)) * NC + ci;
      P[o] = p[s];
      Hend[o] = h[s];
    }
  }
}

__global__ void scan_fixup_kernel(const float* __restrict__ P,
                                  const float* __restrict__ Hend,
                                  float* __restrict__ Hinit, int n, int NC) {
  int i = blockIdx.x * blockDim.x + threadIdx.x;
  if (i >= n) return;
  float carry = 0.f;
  for (int c = 0; c < NC; ++c) {
    long o = (long)i * NC + c;
    Hinit[o] = carry;
    carry = P[o] * carry + Hend[o];
  }
}

// f32 -> bf16 conversion (weights)
__global__ void cvt_kernel(const float* __restrict__ in,
                           unsigned short* __restrict__ out, long n) {
  long i = (long)blockIdx.x * blockDim.x + threadIdx.x;
  for (; i < n; i += (long)gridDim.x * blockDim.x) out[i] = bf_from_f32(in[i]);
}

// f32 (N x K) -> bf16 (N x Kp) with zero pad in K
__global__ void cvt_pad_kernel(const float* __restrict__ in,
                               unsigned short* __restrict__ out,
                               int N, int K, int Kp) {
  long i = (long)blockIdx.x * blockDim.x + threadIdx.x;
  long total = (long)N * Kp;
  if (i >= total) return;
  int n = (int)(i / Kp);
  int k = (int)(i % Kp);
  out[i] = (k < K) ? bf_from_f32(in[(long)n * K + k]) : (unsigned short)0;
}

// ---------------------------------------------------------------------------
// Host orchestration
// ---------------------------------------------------------------------------
extern "C" void kernel_launch(void* const* d_in, const int* in_sizes, int n_in,
                              void* d_out, int out_size, void* d_ws, size_t ws_size,
                              hipStream_t stream) {
  (void)in_sizes; (void)n_in; (void)out_size; (void)ws_size;
  const int d_[4]   = {64, 128, 256, 512};
  const int S_[4]   = {32, 16, 8, 4};
  const int L_[4]   = {32768, 4096, 512, 64};
  const int di_[4]  = {128, 256, 512, 1024};
  const int dtr_[4] = {4, 8, 16, 32};
  const int f_[4]   = {36, 40, 48, 64};
  const int fp_[4]  = {40, 40, 48, 64};  // padded xdbl row stride (mult of 8)

  // ---- d_in index map (JAX pytree flatten: keys sorted at every level) ----
  int ii = 0;
  ii++;  // 'cnn'
  int idx_down_b[3], idx_down_w[3];
  for (int j = 0; j < 3; ++j) { idx_down_b[j] = ii++; idx_down_w[j] = ii++; }
  int idx_mb1[4], idx_mb2[4], idx_mw1[4], idx_mw2[4];
  for (int i = 0; i < 4; ++i) { idx_mb1[i] = ii++; idx_mb2[i] = ii++;
                                idx_mw1[i] = ii++; idx_mw2[i] = ii++; }
  struct LIdx { int A_log, D, conv_b, conv_w, dt_b, dt_w, in_proj, ln_b, ln_w, out_proj, x_proj; };
  LIdx li[4][2];
  for (int i = 0; i < 4; ++i)
    for (int l = 0; l < 2; ++l) {
      li[i][l].A_log = ii++; li[i][l].D = ii++; li[i][l].conv_b = ii++;
      li[i][l].conv_w = ii++; li[i][l].dt_b = ii++; li[i][l].dt_w = ii++;
      li[i][l].in_proj = ii++; li[i][l].ln_b = ii++; li[i][l].ln_w = ii++;
      li[i][l].out_proj = ii++; li[i][l].x_proj = ii++;
    }
  int idx_stem_b = ii++, idx_stem_w = ii++, idx_x = ii++;
  auto IN = [&](int k) { return (const float*)d_in[k]; };

  // ---- workspace bump allocator ----
  char* ws = (char*)d_ws;
  size_t off = 0;
  auto alloc = [&](size_t bytes) -> char* {
    char* p = ws + off;
    off = (off + bytes + 255) & ~(size_t)255;
    return p;
  };
  float* hA   = (float*)alloc(64L * 32768 * 4);
  float* hB   = (float*)alloc(64L * 32768 * 4);
  unsigned short* xln = (unsigned short*)alloc(32768L * 64 * 2);
  float* xz   = (float*)alloc(32768L * 256 * 4);
  float* xc   = (float*)alloc(32768L * 128 * 4);
  unsigned short* xcb = (unsigned short*)alloc(32768L * 128 * 2);
  float* xdbl = (float*)alloc(32768L * 40 * 4);
  unsigned short* xdblb = (unsigned short*)alloc(32768L * 40 * 2);
  float* dtb  = (float*)alloc(32768L * 128 * 4);
  unsigned short* ybf = (unsigned short*)alloc(32768L * 128 * 2);
  unsigned short* t1  = (unsigned short*)alloc(32768L * 128 * 2);
  float* Pbuf  = (float*)alloc(64L * 256 * 16 * 4);
  float* Hend  = (float*)alloc(64L * 256 * 16 * 4);
  float* Hinit = (float*)alloc(64L * 256 * 16 * 4);
  float* meanb = (float*)alloc(512 * 4);
  float* rstdb = (float*)alloc(512 * 4);

  auto cvt = [&](const float* src, size_t n) -> unsigned short* {
    unsigned short* p = (unsigned short*)alloc(n * 2);
    int blocks = (int)((n + 255) / 256);
    if (blocks > 4096) blocks = 4096;
    cvt_kernel<<<blocks, 256, 0, stream>>>(src, p, (long)n);
    return p;
  };
  auto cvt_pad = [&](const float* src, int N, int K, int Kp) -> unsigned short* {
    unsigned short* p = (unsigned short*)alloc((size_t)N * Kp * 2);
    long n = (long)N * Kp;
    cvt_pad_kernel<<<(int)((n + 255) / 256), 256, 0, stream>>>(src, p, N, K, Kp);
    return p;
  };

  // bf16 weight conversions (GEMM B operands only); dt_w zero-padded to K=32
  unsigned short *w_in[4][2], *w_xp[4][2], *w_dt[4][2], *w_out[4][2], *w_m1[4], *w_m2[4];
  for (int i = 0; i < 4; ++i) {
    int d = d_[i], di = di_[i], dtr = dtr_[i], f = f_[i];
    for (int l = 0; l < 2; ++l) {
      w_in[i][l]  = cvt(IN(li[i][l].in_proj),  (size_t)2 * di * d);
      w_xp[i][l]  = cvt(IN(li[i][l].x_proj),   (size_t)f * di);
      w_dt[i][l]  = cvt_pad(IN(li[i][l].dt_w), di, dtr, 32);
      w_out[i][l] = cvt(IN(li[i][l].out_proj), (size_t)d * di);
    }
    w_m1[i] = cvt(IN(idx_mw1[i]), (size_t)2 * d * d);
    w_m2[i] = cvt(IN(idx_mw2[i]), (size_t)d * 2 * d);
  }
  alloc(65536);  // guard slack for B-tile overreads past the last weight

  auto gemm = [&](const unsigned short* A, int lda, const unsigned short* B,
                  const float* bias, float* Cf, unsigned short* Cb,
                  int M, int N, int K, int ldc, int act, int transC) {
    dim3 g((M + GTM - 1) / GTM, (N + GTN - 1) / GTN);
    gemm_bf16_kernel<<<g, 256, 0, stream>>>(A, lda, B, bias, Cf, Cb,
                                            M, N, K, ldc, act, transC);
  };

  // ---- stem ----
  {
    long n = 64L * 32768;
    stem_kernel<<<(int)((n + 255) / 256), 256, 0, stream>>>(
        IN(idx_x), IN(idx_stem_w), IN(idx_stem_b), hA);
  }
  float* hcur = hA;
  float* hoth = hB;

  long out_off[4];
  out_off[0] = 0;
  out_off[1] = out_off[0] + (long)d_[0] * L_[0];
  out_off[2] = out_off[1] + (long)d_[1] * L_[1];
  out_off[3] = out_off[2] + (long)d_[2] * L_[2];

  for (int i = 0; i < 4; ++i) {
    const int d = d_[i], di = di_[i], dtr = dtr_[i], f = f_[i], fp = fp_[i];
    const int L = L_[i];
    if (i > 0) {
      // meanb/rstdb hold stats of hcur from end of previous stage
      int C1 = d_[i - 1], S1 = S_[i - 1];
      long n = (long)d * L;
      down_conv_kernel<<<(int)((n + 255) / 256), 256, 0, stream>>>(
          hcur, meanb, rstdb, IN(idx_down_w[i - 1]), IN(idx_down_b[i - 1]),
          hoth, C1, S1, d);
      float* t = hcur; hcur = hoth; hoth = t;
    }
    // ---- Mamba layers ----
    for (int l = 0; l < 2; ++l) {
      const LIdx& P_ = li[i][l];
      layernorm_kernel<<<(L + 255) / 256, 256, 0, stream>>>(
          hcur, IN(P_.ln_w), IN(P_.ln_b), xln, d, L);
      // xz = xln @ in_proj^T : (L, 2di)
      gemm(xln, d, w_in[i][l], nullptr, xz, nullptr, L, 2 * di, d, 2 * di, 0, 0);
      // depthwise causal conv + SiLU -> xc (f32 + bf16)
      {
        long n = (long)L * di;
        dwconv_silu_kernel<<<(int)((n + 255) / 256), 256, 0, stream>>>(
            xz, IN(P_.conv_w), IN(P_.conv_b), xc, xcb, L, di);
      }
      // xdbl = xc @ x_proj^T : (L, f) stored with padded stride fp (f32 + bf16)
      gemm(xcb, di, w_xp[i][l], nullptr, xdbl, xdblb, L, f, di, fp, 0, 0);
      // dt = softplus(xdbl[:, :dtr] @ dt_w^T + dt_b) : (L, di); K padded to 32
      gemm(xdblb, fp, w_dt[i][l], IN(P_.dt_b), dtb, nullptr, L, di, 32, di, 1, 0);
      // chunked selective scan
      int NC = (L >= 4096) ? 64 : (L / 64 > 0 ? L / 64 : 1);
      int CL = L / NC;
      int nthr = NC * di;
      scan_pass_kernel<<<(nthr + 255) / 256, 256, 0, stream>>>(
          dtb, xdbl, xc, xz, IN(P_.A_log), IN(P_.D),
          Pbuf, Hend, nullptr, nullptr, L, di, fp, dtr, NC, CL, 1);
      int nes = di * 16;
      scan_fixup_kernel<<<(nes + 255) / 256, 256, 0, stream>>>(
          Pbuf, Hend, Hinit, nes, NC);
      scan_pass_kernel<<<(nthr + 255) / 256, 256, 0, stream>>>(
          dtb, xdbl, xc, xz, IN(P_.A_log), IN(P_.D),
          nullptr, nullptr, Hinit, ybf, L, di, fp, dtr, NC, CL, 2);
      // h' = y @ out_proj^T, transposed store into (C, L)
      gemm(ybf, di, w_out[i][l], nullptr, hoth, nullptr, L, d, di, d, 0, 1);
      float* t = hcur; hcur = hoth; hoth = t;
    }
    // ---- instance-norm stats on hcur (MLP here; down conv next stage) ----
    chan_stats_kernel<<<d, 256, 0, stream>>>(hcur, meanb, rstdb, L);
    {
      long n = (long)d * L;
      in_transpose_kernel<<<(int)((n + 255) / 256), 256, 0, stream>>>(
          hcur, meanb, rstdb, xln, d, L);
    }
    // ---- MLP: t1 = gelu(xln @ w1^T + b1); out = t1 @ w2^T + b2 -> (C,L) ----
    gemm(xln, d, w_m1[i], IN(idx_mb1[i]), nullptr, t1, L, 2 * d, d, 2 * d, 2, 0);
    gemm(t1, 2 * d, w_m2[i], IN(idx_mb2[i]),
         (float*)d_out + out_off[i], nullptr, L, d, 2 * d, d, 0, 1);
  }
}
